// TransformerLM_64347200029404
// MI455X (gfx1250) — matrix operand order repaired and checked by reading, not compiled, not run on hardware
//
#include <hip/hip_runtime.h>
#include <math.h>

// ---------------------------------------------------------------------------
// TransformerLM forward for MI455X (gfx1250, wave32, WMMA bf16 16x16x32)
// ---------------------------------------------------------------------------

typedef __attribute__((ext_vector_type(16))) __bf16 v16bf;
typedef __attribute__((ext_vector_type(8)))  __bf16 v8bf;
typedef __attribute__((ext_vector_type(8)))  float  v8f;
typedef int v4i __attribute__((vector_size(16)));

#define B_  2
#define S_  1024
#define D_  1024
#define H_  16
#define DK_ 64
#define F_  4096
#define L_  6
#define V_  32000

enum { MODE_QT = 0, MODE_KV = 1, MODE_RESID_F32 = 2, MODE_GELU_BF16 = 3, MODE_PLAIN_F32 = 4 };

// ---- CDNA5 async global->LDS copy (ASYNCcnt) with safe fallback -----------
#if defined(__has_builtin)
#if __has_builtin(__builtin_amdgcn_global_load_async_to_lds_b128)
#define HAVE_ASYNC_LDS 1
#endif
#if __has_builtin(__builtin_amdgcn_s_wait_asynccnt)
#define HAVE_WAIT_ASYNC_BUILTIN 1
#endif
#endif

#if defined(HAVE_WAIT_ASYNC_BUILTIN)
#define WAIT_ASYNC(n) __builtin_amdgcn_s_wait_asynccnt(n)
#elif defined(HAVE_ASYNC_LDS)
#define WAIT_ASYNC(n) asm volatile("s_wait_asynccnt %0" ::"n"(n) : "memory")
#else
#define WAIT_ASYNC(n) ((void)0)
#endif

// exact param types per compiler diagnostic: (as1 int4*, as3 int4*, Ii, Ii)
#define GPTR_V4I(p) ((__attribute__((address_space(1))) v4i*)(v4i*)(p))
#define LPTR_V4I(p) ((__attribute__((address_space(3))) v4i*)(v4i*)(p))

// ---------------------------------------------------------------------------
// fp32 -> bf16 weight conversion (grid-stride)
// ---------------------------------------------------------------------------
__global__ void cvt_bf16_kernel(const float* __restrict__ in, __bf16* __restrict__ out, size_t n) {
  size_t i = (size_t)blockIdx.x * blockDim.x + threadIdx.x;
  size_t stride = (size_t)gridDim.x * blockDim.x;
  for (; i < n; i += stride) out[i] = (__bf16)in[i];
}

// emb [V,D] fp32 -> embT [D,V] bf16 (tiled transpose)
__global__ __launch_bounds__(256) void transpose_embT_kernel(const float* __restrict__ emb,
                                                             __bf16* __restrict__ embT) {
  __shared__ float tile[32][33];
  const int v0 = blockIdx.x * 32, d0 = blockIdx.y * 32;
  const int tx = threadIdx.x & 31, ty = threadIdx.x >> 5;   // 32 x 8
  #pragma unroll
  for (int i = 0; i < 4; ++i) {
    int r = ty + i * 8;
    tile[r][tx] = emb[(size_t)(v0 + r) * D_ + d0 + tx];
  }
  __syncthreads();
  #pragma unroll
  for (int i = 0; i < 4; ++i) {
    int r = ty + i * 8;
    embT[(size_t)(d0 + r) * V_ + v0 + tx] = (__bf16)tile[tx][r];
  }
}

// ---------------------------------------------------------------------------
// Embedding + sinusoidal positional encoding -> x fp32 [B,S,D]
// ---------------------------------------------------------------------------
__global__ __launch_bounds__(256) void embed_kernel(const int* __restrict__ ids,
                                                    const float* __restrict__ emb,
                                                    float* __restrict__ x) {
  const int s = blockIdx.x, b = blockIdx.y;
  const int id = ids[b * S_ + s];
  const float sqrtD = 32.0f;                        // sqrt(1024)
  const float nl = -9.210340371976184f / (float)D_; // -ln(10000)/D
  #pragma unroll
  for (int i = 0; i < 4; ++i) {
    int d = threadIdx.x + i * 256;
    int p = d >> 1;
    float div = __expf((float)(2 * p) * nl);
    float ang = (float)s * div;
    float pe = (d & 1) ? __cosf(ang) : __sinf(ang);
    x[((size_t)(b * S_ + s)) * D_ + d] = emb[(size_t)id * D_ + d] * sqrtD + pe;
  }
}

// ---------------------------------------------------------------------------
// LayerNorm: fp32 row in, bf16 row out
// ---------------------------------------------------------------------------
__global__ __launch_bounds__(256) void layernorm_kernel(const float* __restrict__ x,
                                                        const float* __restrict__ g,
                                                        const float* __restrict__ be,
                                                        __bf16* __restrict__ out) {
  const int row = blockIdx.x;
  const int tid = threadIdx.x;
  const float* xr = x + (size_t)row * D_;
  __shared__ float red[256];

  float vals[4];
  float s = 0.f;
  #pragma unroll
  for (int i = 0; i < 4; ++i) { vals[i] = xr[tid + i * 256]; s += vals[i]; }
  red[tid] = s; __syncthreads();
  for (int off = 128; off > 0; off >>= 1) {
    if (tid < off) red[tid] += red[tid + off];
    __syncthreads();
  }
  const float mean = red[0] * (1.0f / (float)D_);
  __syncthreads();

  float v = 0.f;
  #pragma unroll
  for (int i = 0; i < 4; ++i) { float d = vals[i] - mean; v += d * d; }
  red[tid] = v; __syncthreads();
  for (int off = 128; off > 0; off >>= 1) {
    if (tid < off) red[tid] += red[tid + off];
    __syncthreads();
  }
  const float inv = rsqrtf(red[0] * (1.0f / (float)D_) + 1e-5f);

  #pragma unroll
  for (int i = 0; i < 4; ++i) {
    int d = tid + i * 256;
    out[(size_t)row * D_ + d] = (__bf16)((vals[i] - mean) * inv * g[d] + be[d]);
  }
}

// ---------------------------------------------------------------------------
// WMMA GEMM: C[M,N] = A[M,K](bf16,row) x Bw[K,N](bf16,row) + bias, fused epilogue.
// workgroup = 256 threads = 8 waves; wave computes a 16(M) x 64(N) tile.
// B tile (32K x 64N) is staged in LDS via double-buffered async copies so all
// 8 waves share one global fetch; copy of step i+1 overlaps WMMAs of step i.
// ---------------------------------------------------------------------------
template<int MODE>
__global__ __launch_bounds__(256) void gemm_wmma(const __bf16* __restrict__ A,
                                                 const __bf16* __restrict__ Bw,
                                                 const float* __restrict__ bias,
                                                 const float* resid,
                                                 void* outp, int M, int N, int K) {
  const int lane = threadIdx.x & 31;
  const int wave = threadIdx.x >> 5;
  const int lm = lane & 15;   // A row within tile / output column
  const int lh = lane >> 4;   // A k-chunk group / output row-half
  const int m0 = (blockIdx.y * 8 + wave) * 16;
  const int n0 = blockIdx.x * 64;

  // B staging buffer: 2 x 32 rows x 144B (72 bf16) padded stride:
  // 16B-aligned rows for ds_load_b128, worst-case 2-way bank conflict.
  __shared__ __align__(16) __bf16 sB[2][32][72];

  const int tid = threadIdx.x;
  const int srow = tid >> 3;        // 0..31  (k row within tile)
  const int sseg = tid & 7;         // 0..7   (16B segment within row)

  auto stageB = [&](int buf, int k0) {
    const __bf16* src = Bw + (size_t)(k0 + srow) * N + n0 + sseg * 8;
    __bf16* dst = &sB[buf][srow][sseg * 8];
#if defined(HAVE_ASYNC_LDS)
    __builtin_amdgcn_global_load_async_to_lds_b128(GPTR_V4I(src), LPTR_V4I(dst), 0, 0);
#else
    *(v8bf*)dst = *(const v8bf*)src;
#endif
  };

  v8f acc[4] = {};
  const __bf16* arow = A + (size_t)(m0 + lm) * K + lh * 8;
  const int nsteps = K >> 5;

  stageB(0, 0);
  for (int i = 0; i < nsteps; ++i) {
    const int cur = i & 1;
    const int k0 = i * 32;
    if (i + 1 < nsteps) {
      stageB(cur ^ 1, k0 + 32);
      WAIT_ASYNC(1);                 // own copy for buf[cur] complete
    } else {
      WAIT_ASYNC(0);
    }
    __syncthreads();                 // every wave's rows of buf[cur] present

    if (k0 + 96 < K) __builtin_prefetch(arow + k0 + 96, 0, 1);
    v8bf c0 = *(const v8bf*)(arow + k0);
    v8bf c1 = *(const v8bf*)(arow + k0 + 16);
    v16bf af = __builtin_shufflevector(c0, c1, 0,1,2,3,4,5,6,7,8,9,10,11,12,13,14,15);
    #pragma unroll
    for (int t = 0; t < 4; ++t) {
      v16bf bf;
      #pragma unroll
      for (int s = 0; s < 16; ++s) bf[s] = sB[cur][(s < 8 ? s : s + 8) + lh * 8][t * 16 + lm];
      acc[t] = __builtin_amdgcn_wmma_f32_16x16x32_bf16(false, af, false, bf,
                                                       (short)0, acc[t], false, false);
    }
    __syncthreads();                 // all reads of buf[cur] done before overwrite
  }

  #pragma unroll
  for (int t = 0; t < 4; ++t) {
    #pragma unroll
    for (int r = 0; r < 8; ++r) {
      const int mg = m0 + r + lh * 8;
      const int ng = n0 + t * 16 + lm;
      float v = acc[t][r];
      if (bias) v += bias[ng];
      if (MODE == MODE_RESID_F32) {
        v += resid[(size_t)mg * N + ng];
        ((float*)outp)[(size_t)mg * N + ng] = v;
      } else if (MODE == MODE_PLAIN_F32) {
        ((float*)outp)[(size_t)mg * N + ng] = v;
      } else if (MODE == MODE_GELU_BF16) {
        float gl = 0.5f * v * (1.0f + erff(v * 0.70710678118654752f));
        ((__bf16*)outp)[(size_t)mg * N + ng] = (__bf16)gl;
      } else if (MODE == MODE_QT) {          // q^T: [B,H,DK,S]
        int b = mg >> 10, s = mg & (S_ - 1);
        int hh = ng >> 6, d = ng & (DK_ - 1);
        ((__bf16*)outp)[((size_t)(b * H_ + hh) * DK_ + d) * S_ + s] = (__bf16)v;
      } else {                                // k/v: [B,H,S,DK]
        int b = mg >> 10, s = mg & (S_ - 1);
        int hh = ng >> 6, d = ng & (DK_ - 1);
        ((__bf16*)outp)[((size_t)(b * H_ + hh) * S_ + s) * DK_ + d] = (__bf16)v;
      }
    }
  }
}

// ---------------------------------------------------------------------------
// Flash-style causal attention with WMMA.
// Per wave: one 16-query block of one (b,h). Two passes over key blocks of 32.
// S^T = K x Q^T (both operands contiguous thanks to q^T layout); per-query
// softmax stats live per-lane; the S^T D-fragment pair maps exactly onto the
// A-operand layout of P[16q x 32k] for O = P x V.
// ---------------------------------------------------------------------------
__global__ __launch_bounds__(128) void attn_kernel(const __bf16* __restrict__ qt,
                                                   const __bf16* __restrict__ kk,
                                                   const __bf16* __restrict__ vv,
                                                   __bf16* __restrict__ attn) {
  const int lane = threadIdx.x & 31;
  const int wave = threadIdx.x >> 5;
  const int qb = blockIdx.x * 4 + wave;      // 0..63
  const int h = blockIdx.y, b = blockIdx.z;
  const int q0 = qb * 16;
  const int lm = lane & 15, lh = lane >> 4;
  const float scale = 0.125f;                // 1/sqrt(64)
  const float NEG = -3.0e38f;

  const __bf16* qtp = qt + (size_t)(b * H_ + h) * DK_ * S_;
  const __bf16* kp  = kk + (size_t)(b * H_ + h) * S_ * DK_;
  const __bf16* vp  = vv + (size_t)(b * H_ + h) * S_ * DK_;

  // Q^T B-fragments: lane%16 = query column, elements = dims in table k order
  v16bf qf0, qf1;
  #pragma unroll
  for (int s = 0; s < 16; ++s) {
    const int kr = (s < 8 ? s : s + 8) + (lane >> 4) * 8;
    qf0[s] = qtp[(size_t)kr * S_ + q0 + (lane & 15)];
    qf1[s] = qtp[(size_t)(32 + kr) * S_ + q0 + (lane & 15)];
  }

  const int qg = q0 + lm;                    // this lane's query column
  const int nkb = (q0 + 16 + 31) / 32;       // 32-key blocks covering causal span

  // ---- pass 1: per-query max & sum (online, per lane) ----
  float mrow = NEG, srow = 0.f;
  for (int kb = 0; kb < nkb; ++kb) {
    #pragma unroll
    for (int half = 0; half < 2; ++half) {
      const int kbase = kb * 32 + half * 16;
      v8f sacc = {};
      #pragma unroll
      for (int ks = 0; ks < 2; ++ks) {
        const __bf16* ar = kp + (size_t)(kbase + lm) * DK_ + ks * 32 + lh * 8;
        v8bf c0 = *(const v8bf*)ar;
        v8bf c1 = *(const v8bf*)(ar + 16);
        v16bf af = __builtin_shufflevector(c0, c1, 0,1,2,3,4,5,6,7,8,9,10,11,12,13,14,15);
        sacc = __builtin_amdgcn_wmma_f32_16x16x32_bf16(false, af, false, (ks ? qf1 : qf0),
                                                       (short)0, sacc, false, false);
      }
      float tv[8], tmax = NEG;
      #pragma unroll
      for (int r = 0; r < 8; ++r) {
        int kg = kbase + r + lh * 8;
        float t = (kg <= qg) ? sacc[r] * scale : NEG;
        tv[r] = t;
        tmax = fmaxf(tmax, t);
      }
      float mnew = fmaxf(mrow, tmax);
      float acc = 0.f;
      #pragma unroll
      for (int r = 0; r < 8; ++r) acc += __expf(tv[r] - mnew);
      srow = srow * __expf(mrow - mnew) + acc;
      mrow = mnew;
    }
  }
  // combine the two lane-halves holding the same query column
  {
    float m_o = __shfl_xor(mrow, 16, 32);
    float s_o = __shfl_xor(srow, 16, 32);
    float M = fmaxf(mrow, m_o);
    srow = srow * __expf(mrow - M) + s_o * __expf(m_o - M);
    mrow = M;
  }
  const float inv = 1.0f / srow;

  // ---- pass 2: P = exp(s - m)/sum, O += P x V ----
  v8f oacc[4] = {};
  for (int kb = 0; kb < nkb; ++kb) {
    v8bf ph[2];
    #pragma unroll
    for (int half = 0; half < 2; ++half) {
      const int kbase = kb * 32 + half * 16;
      v8f sacc = {};
      #pragma unroll
      for (int ks = 0; ks < 2; ++ks) {
        const __bf16* ar = kp + (size_t)(kbase + lm) * DK_ + ks * 32 + lh * 8;
        v8bf c0 = *(const v8bf*)ar;
        v8bf c1 = *(const v8bf*)(ar + 16);
        v16bf af = __builtin_shufflevector(c0, c1, 0,1,2,3,4,5,6,7,8,9,10,11,12,13,14,15);
        sacc = __builtin_amdgcn_wmma_f32_16x16x32_bf16(false, af, false, (ks ? qf1 : qf0),
                                                       (short)0, sacc, false, false);
      }
      v8bf p;
      #pragma unroll
      for (int r = 0; r < 8; ++r) {
        int kg = kbase + r + lh * 8;
        float t = (kg <= qg) ? sacc[r] * scale : NEG;
        p[r] = (__bf16)(__expf(t - mrow) * inv);
      }
      ph[half] = p;
    }
    // P A-fragment over 32 keys: elements 0..7 <- half0, 8..15 <- half1
    v16bf pa = __builtin_shufflevector(ph[0], ph[1], 0,1,2,3,4,5,6,7,8,9,10,11,12,13,14,15);
    #pragma unroll
    for (int dt = 0; dt < 4; ++dt) {
      v16bf bv;
      #pragma unroll
      for (int s = 0; s < 16; ++s) bv[s] = vp[(size_t)(kb * 32 + (s < 8 ? s : s + 8) + lh * 8) * DK_ + dt * 16 + lm];
      oacc[dt] = __builtin_amdgcn_wmma_f32_16x16x32_bf16(false, pa, false, bv,
                                                         (short)0, oacc[dt], false, false);
    }
  }

  // store O tile -> attn [B,S,D] bf16 (row-major, head-sliced)
  __bf16* op = attn + (size_t)(b * S_ + q0) * D_ + h * DK_;
  #pragma unroll
  for (int dt = 0; dt < 4; ++dt) {
    #pragma unroll
    for (int r = 0; r < 8; ++r) {
      int ql = r + lh * 8;
      op[(size_t)ql * D_ + dt * 16 + lm] = (__bf16)oacc[dt][r];
    }
  }
}

// ---------------------------------------------------------------------------
// Host driver
// ---------------------------------------------------------------------------
extern "C" void kernel_launch(void* const* d_in, const int* in_sizes, int n_in,
                              void* d_out, int out_size, void* d_ws, size_t ws_size,
                              hipStream_t stream) {
  (void)in_sizes; (void)n_in; (void)out_size; (void)ws_size;

  const int*   ids   = (const int*)d_in[0];
  const float* emb   = (const float*)d_in[1];
  const float* wq    = (const float*)d_in[2];
  const float* bq    = (const float*)d_in[3];
  const float* wk    = (const float*)d_in[4];
  const float* bk    = (const float*)d_in[5];
  const float* wv    = (const float*)d_in[6];
  const float* bv    = (const float*)d_in[7];
  const float* wo    = (const float*)d_in[8];
  const float* bo    = (const float*)d_in[9];
  const float* ln1g  = (const float*)d_in[10];
  const float* ln1b  = (const float*)d_in[11];
  const float* ln2g  = (const float*)d_in[12];
  const float* ln2b  = (const float*)d_in[13];
  const float* w1    = (const float*)d_in[14];
  const float* b1    = (const float*)d_in[15];
  const float* w2    = (const float*)d_in[16];
  const float* b2    = (const float*)d_in[17];
  const float* lnfg  = (const float*)d_in[18];
  const float* lnfb  = (const float*)d_in[19];

  // workspace carving (256B aligned)
  char* p = (char*)d_ws;
  auto carve = [&](size_t bytes) -> void* {
    void* r = (void*)p;
    p += (bytes + 255) & ~(size_t)255;
    return r;
  };
  const size_t nDD = (size_t)L_ * D_ * D_;
  const size_t nDF = (size_t)L_ * D_ * F_;
  __bf16* wq_bf  = (__bf16*)carve(nDD * 2);
  __bf16* wk_bf  = (__bf16*)carve(nDD * 2);
  __bf16* wv_bf  = (__bf16*)carve(nDD * 2);
  __bf16* wo_bf  = (__bf16*)carve(nDD * 2);
  __bf16* w1_bf  = (__bf16*)carve(nDF * 2);
  __bf16* w2_bf  = (__bf16*)carve(nDF * 2);
  __bf16* embT   = (__bf16*)carve((size_t)D_ * V_ * 2);
  float*  x      = (float*) carve((size_t)B_ * S_ * D_ * 4);
  __bf16* hbuf   = (__bf16*)carve((size_t)B_ * S_ * D_ * 2);
  __bf16* qt     = (__bf16*)carve((size_t)B_ * S_ * D_ * 2);
  __bf16* kbuf   = (__bf16*)carve((size_t)B_ * S_ * D_ * 2);
  __bf16* vbuf   = (__bf16*)carve((size_t)B_ * S_ * D_ * 2);
  __bf16* attnb  = (__bf16*)carve((size_t)B_ * S_ * D_ * 2);
  __bf16* ffnb   = (__bf16*)carve((size_t)B_ * S_ * F_ * 2);

  const int M = B_ * S_;   // 2048 rows

  // 1) weight precision conversion + emb transpose
  cvt_bf16_kernel<<<2048, 256, 0, stream>>>(wq, wq_bf, nDD);
  cvt_bf16_kernel<<<2048, 256, 0, stream>>>(wk, wk_bf, nDD);
  cvt_bf16_kernel<<<2048, 256, 0, stream>>>(wv, wv_bf, nDD);
  cvt_bf16_kernel<<<2048, 256, 0, stream>>>(wo, wo_bf, nDD);
  cvt_bf16_kernel<<<4096, 256, 0, stream>>>(w1, w1_bf, nDF);
  cvt_bf16_kernel<<<4096, 256, 0, stream>>>(w2, w2_bf, nDF);
  transpose_embT_kernel<<<dim3(V_ / 32, D_ / 32), 256, 0, stream>>>(emb, embT);

  // 2) embedding + positional encoding
  embed_kernel<<<dim3(S_, B_), 256, 0, stream>>>(ids, emb, x);

  // 3) layers
  for (int l = 0; l < L_; ++l) {
    const __bf16* wq_l = wq_bf + (size_t)l * D_ * D_;
    const __bf16* wk_l = wk_bf + (size_t)l * D_ * D_;
    const __bf16* wv_l = wv_bf + (size_t)l * D_ * D_;
    const __bf16* wo_l = wo_bf + (size_t)l * D_ * D_;
    const __bf16* w1_l = w1_bf + (size_t)l * D_ * F_;
    const __bf16* w2_l = w2_bf + (size_t)l * F_ * D_;
    const float* bq_l = bq + (size_t)l * D_;
    const float* bk_l = bk + (size_t)l * D_;
    const float* bv_l = bv + (size_t)l * D_;
    const float* bo_l = bo + (size_t)l * D_;
    const float* b1_l = b1 + (size_t)l * F_;
    const float* b2_l = b2 + (size_t)l * D_;

    // LN1
    layernorm_kernel<<<M, 256, 0, stream>>>(x, ln1g + (size_t)l * D_, ln1b + (size_t)l * D_, hbuf);
    // Q/K/V projections with layout-fused epilogues
    gemm_wmma<MODE_QT><<<dim3(D_ / 64, M / 128), 256, 0, stream>>>(hbuf, wq_l, bq_l, nullptr, qt,   M, D_, D_);
    gemm_wmma<MODE_KV><<<dim3(D_ / 64, M / 128), 256, 0, stream>>>(hbuf, wk_l, bk_l, nullptr, kbuf, M, D_, D_);
    gemm_wmma<MODE_KV><<<dim3(D_ / 64, M / 128), 256, 0, stream>>>(hbuf, wv_l, bv_l, nullptr, vbuf, M, D_, D_);
    // causal attention
    attn_kernel<<<dim3(S_ / 64, H_, B_), 128, 0, stream>>>(qt, kbuf, vbuf, attnb);
    // output projection + residual (in-place into x)
    gemm_wmma<MODE_RESID_F32><<<dim3(D_ / 64, M / 128), 256, 0, stream>>>(attnb, wo_l, bo_l, x, x, M, D_, D_);
    // LN2 + FFN
    layernorm_kernel<<<M, 256, 0, stream>>>(x, ln2g + (size_t)l * D_, ln2b + (size_t)l * D_, hbuf);
    gemm_wmma<MODE_GELU_BF16><<<dim3(F_ / 64, M / 128), 256, 0, stream>>>(hbuf, w1_l, b1_l, nullptr, ffnb, M, F_, D_);
    gemm_wmma<MODE_RESID_F32><<<dim3(D_ / 64, M / 128), 256, 0, stream>>>(ffnb, w2_l, b2_l, x, x, M, D_, F_);
  }

  // 4) final LN + tied lm_head logits (fp32 out)
  layernorm_kernel<<<M, 256, 0, stream>>>(x, lnfg, lnfb, hbuf);
  gemm_wmma<MODE_PLAIN_F32><<<dim3(V_ / 64, M / 128), 256, 0, stream>>>(hbuf, embT, nullptr, nullptr,
                                                                        d_out, M, V_, D_);
}
